// ALFEblock_28226525069393
// MI455X (gfx1250) — matrix operand order repaired and checked by reading, not compiled, not run on hardware
//
#include <hip/hip_runtime.h>
#include <hip/hip_fp16.h>
#include <math.h>

// Problem constants (B,C,H,W) = (4,64,128,128)
#define B_  4
#define C_  64
#define H_  128
#define W_  128
#define HH_ 64
#define WH_ 64
#define N_  4096   // HH_*WH_

typedef __attribute__((ext_vector_type(16))) _Float16 v16h;
typedef __attribute__((ext_vector_type(8)))  _Float16 v8h;
typedef __attribute__((ext_vector_type(8)))  float    v8f;

// ---------------- WMMA fragment helpers (wave32, 16x16x32 f16) ----------------
// A fragment: 16(M) x 32(K), tile row-major with row stride `ld` (elements).
// lane l: row = l&15; K halves at 8*(l>>4) and 16+8*(l>>4)  (matches ISA layout).
__device__ __forceinline__ v16h load_frag_a(const _Float16* t, int ld) {
  const int l = threadIdx.x & 31;
  const int r = l & 15, hi = l >> 4;
  const _Float16* p = t + (size_t)r * ld + hi * 8;
  v8h lo = *(const v8h*)(p);
  v8h hv = *(const v8h*)(p + 16);
  v16h o;
#pragma unroll
  for (int i = 0; i < 8; ++i) { o[i] = lo[i]; o[i + 8] = hv[i]; }
  return o;
}

// B fragment: 32(K) x 16(N); `bt` points at B-transpose (N-major, row stride ld).
// lane l: column n = l&15; K halves at 8*(l>>4) and 16+8*(l>>4) (matches A).
__device__ __forceinline__ v16h load_frag_b(const _Float16* bt, int ld) {
  const int l = threadIdx.x & 31;
  const int n = l & 15, hi = l >> 4;
  const _Float16* p = bt + (size_t)n * ld + hi * 8;
  v8h a = *(const v8h*)(p);
  v8h b = *(const v8h*)(p + 16);
  v16h o;
#pragma unroll
  for (int i = 0; i < 8; ++i) { o[i] = a[i]; o[i + 8] = b[i]; }
  return o;
}

__device__ __forceinline__ v8f wmma32(v16h a, v16h b, v8f c) {
  // D = A*B + C ; fp32 accumulate
  return __builtin_amdgcn_wmma_f32_16x16x32_f16(false, a, false, b, (short)0, c,
                                                false, false);
}
// D/C layout: lane l, vgpr r -> element (row = r + 8*(l>>4), col = l&15)

// CDNA5 async copy: per-lane 16B global -> LDS, tracked by ASYNCcnt.
// lds_off = wave-relative LDS byte address (low 32 bits of generic pointer).
__device__ __forceinline__ void async_b128(unsigned lds_off, const void* g) {
  asm volatile("global_load_async_to_lds_b128 %0, %1, off"
               :: "v"(lds_off), "v"(g) : "memory");
}

// ---------------- Kernel 0: weights fp32 -> f16 ----------------
__global__ void alfe_cvt_weights(const float* w2, const float* w3, const float* w4,
                                 const float* w5, const float* w6, const float* w1,
                                 _Float16* wh5, _Float16* w1h) {
  int i = blockIdx.x * 256 + threadIdx.x;
  if (i < 4096) {
    wh5[0 * 4096 + i] = (_Float16)w2[i];
    wh5[1 * 4096 + i] = (_Float16)w3[i];
    wh5[2 * 4096 + i] = (_Float16)w4[i];
    wh5[3 * 4096 + i] = (_Float16)w5[i];
    wh5[4 * 4096 + i] = (_Float16)w6[i];
  }
  if (i < 8192) w1h[i] = (_Float16)w1[i];
}

// ---------------- Kernel 1: pixel-unshuffle + LayerNorm(256) + 256->64 proj ----
// Block: 256 threads, handles 32 consecutive spatial positions of one batch.
__global__ __launch_bounds__(256)
void alfe_unshuffle_ln_proj(const float* __restrict__ x,
                            const float* __restrict__ gamma,
                            const float* __restrict__ beta,
                            const float* __restrict__ pmw,   // (64,256) row-major
                            const float* __restrict__ pmb,
                            _Float16* __restrict__ c1h) {    // (B,N,64) f16
  const int blk = blockIdx.x;
  const int b   = blk / (N_ / 32);
  const int n0  = (blk % (N_ / 32)) * 32;
  const int hh  = n0 / WH_;
  const int ww0 = n0 % WH_;
  const int tid = threadIdx.x;

  __shared__ float tile[256][33];   // [channel][pos], pad 33 (gcd(33,64)=1)
  __shared__ float ps[8][32], pss[8][32], mu[32], rs[32];

  // channel mapping: group g in {x0,x1_,x2_,x3_} -> (dh,dw) = (g&1, g>>1)
  const int g = tid >> 6, c = tid & 63;
  const int dh = g & 1, dw = g >> 1;
  const float* xb = x + (((size_t)b * C_ + c) * H_ + (2 * hh + dh)) * W_ + dw;
#pragma unroll 4
  for (int p = 0; p < 32; ++p) tile[tid][p] = xb[2 * (ww0 + p)];
  __syncthreads();

  // partial sums: 8 segments of 32 channels per position
  {
    const int pos = tid & 31, seg = tid >> 5;
    float s = 0.f, ss = 0.f;
    for (int k = seg * 32; k < seg * 32 + 32; ++k) {
      float v = tile[k][pos]; s += v; ss += v * v;
    }
    ps[seg][pos] = s; pss[seg][pos] = ss;
  }
  __syncthreads();
  if (tid < 32) {
    float s = 0.f, ss = 0.f;
    for (int q = 0; q < 8; ++q) { s += ps[q][tid]; ss += pss[q][tid]; }
    float m = s * (1.f / 256.f);
    float var = ss * (1.f / 256.f) - m * m;
    mu[tid] = m; rs[tid] = rsqrtf(var + 1e-5f);
  }
  __syncthreads();
  {
    const float gm = gamma[tid], bt = beta[tid];
    for (int p = 0; p < 32; ++p)
      tile[tid][p] = (tile[tid][p] - mu[p]) * rs[p] * gm + bt;
  }
  __syncthreads();

  // t @ pm_w.T : 32 pos x 64 cout, 8 outputs/thread (LDS reads broadcast per wave)
  const int co = tid & 63, pbase = (tid >> 6) * 8;
  const float* wrow = pmw + (size_t)co * 256;
  for (int pp = 0; pp < 8; ++pp) {
    const int p = pbase + pp;
    float acc = pmb[co];
#pragma unroll 8
    for (int k = 0; k < 256; ++k) acc += tile[k][p] * wrow[k];
    c1h[((size_t)b * N_ + n0 + p) * C_ + co] = (_Float16)acc;
  }
}

// ---------------- Kernel 2: five C x C projections via WMMA ----------------
// One wave per (b, 16-row n-tile). Shared A-frags, 5 proj x 4 cout-tiles x 2 k-steps.
__global__ __launch_bounds__(32)
void alfe_qkv_proj(const _Float16* __restrict__ c1h,   // (B,N,64)
                   const _Float16* __restrict__ wh5,   // w2..w6, each (64,64)
                   const float* __restrict__ b2, const float* __restrict__ b3,
                   const float* __restrict__ b4, const float* __restrict__ b5,
                   const float* __restrict__ b6,
                   _Float16* __restrict__ qnc,  _Float16* __restrict__ knc,
                   _Float16* __restrict__ vcn,  _Float16* __restrict__ vnc,
                   _Float16* __restrict__ q2cn, _Float16* __restrict__ k2cn) {
  const int b = blockIdx.y;
  const int n0 = blockIdx.x * 16;
  const _Float16* ab = c1h + ((size_t)b * N_ + n0) * C_;
  const v16h a0 = load_frag_a(ab, C_);
  const v16h a1 = load_frag_a(ab + 32, C_);
  const int l = threadIdx.x & 31, hi = l >> 4, col = l & 15;
  const float* biases[5] = {b2, b3, b4, b5, b6};
#pragma unroll
  for (int p = 0; p < 5; ++p) {
    const _Float16* wp = wh5 + (size_t)p * 4096;  // B^T == w (cout,cin) row-major
#pragma unroll
    for (int ct = 0; ct < 4; ++ct) {
      v8f acc{};
      acc = wmma32(a0, load_frag_b(wp + ct * 16 * C_, C_), acc);
      acc = wmma32(a1, load_frag_b(wp + ct * 16 * C_ + 32, C_), acc);
      const float bias = biases[p][ct * 16 + col];
#pragma unroll
      for (int r = 0; r < 8; ++r) {
        const int n = n0 + r + 8 * hi;
        const int c = ct * 16 + col;
        const _Float16 hv = (_Float16)(acc[r] + bias);
        const size_t inc = ((size_t)b * N_ + n) * C_ + c;         // (N,C)
        const size_t icn = ((size_t)b * C_ + c) * (size_t)N_ + n; // (C,N)
        if (p == 0)      qnc[inc] = hv;
        else if (p == 1) knc[inc] = hv;
        else if (p == 2) { vcn[icn] = hv; vnc[inc] = hv; }
        else if (p == 3) q2cn[icn] = hv;
        else             k2cn[icn] = hv;
      }
    }
  }
}

// ---------------- Kernel 3: spatial flash attention (N=4096 keys) ------------
// Block = 8 waves (256 threads); each wave owns 16 queries, all waves share a
// 32-key K-tile (32x64 f16) + V-tile (64x32 f16) staged into LDS with
// double-buffered GLOBAL_LOAD_ASYNC_TO_LDS_B128 (ASYNCcnt). Per 32-key step:
// 4 WMMA for S (B-frags from LDS), online softmax via shfl_xor, P relayout
// through a per-wave LDS slab, 4 WMMA for O += P * V^T.
#define FA_WAVES 8
__global__ __launch_bounds__(256)
void alfe_flash_attn(const _Float16* __restrict__ qnc,  // (B,N,64)
                     const _Float16* __restrict__ knc,  // (B,N,64)
                     const _Float16* __restrict__ vcn,  // (B,64,N)
                     _Float16* __restrict__ out1nc) {   // (B,N,64)
  const int b   = blockIdx.y;
  const int tid = threadIdx.x;
  const int wid = tid >> 5;
  const int l   = tid & 31, hi = l >> 4, col = l & 15;
  const int q0  = blockIdx.x * (16 * FA_WAVES) + wid * 16;

  __shared__ __align__(16) _Float16 Klds[2][32 * 64];        // [key][ch], ld=64
  __shared__ __align__(16) _Float16 Vlds[2][64 * 32];        // [ch][key], ld=32
  __shared__ __align__(16) _Float16 Plds[FA_WAVES][16 * 32]; // per-wave P, ld=32

  const _Float16* qb = qnc + ((size_t)b * N_ + q0) * C_;
  const v16h qa0 = load_frag_a(qb, C_);
  const v16h qa1 = load_frag_a(qb + 32, C_);

  const _Float16* kbase = knc + (size_t)b * N_ * C_;            // rows contiguous
  const _Float16* vbase = vcn + (size_t)b * C_ * (size_t)N_;
  const int vrow = tid >> 2, vseg = tid & 3;                    // V staging map

  v8f O0{}, O1{}, O2{}, O3{};
  float mrow[8], lrow[8];
#pragma unroll
  for (int r = 0; r < 8; ++r) { mrow[r] = -1e30f; lrow[r] = 0.f; }

  // ---- stage key-block kb into buffer ibuf (2 async b128 per thread) ----
  auto stage = [&](int kb, int ibuf) {
    // K tile: 32 rows x 128B == one contiguous 4KB block
    async_b128((unsigned)(size_t)&Klds[ibuf][0] + tid * 16,
               (const char*)(kbase + (size_t)kb * C_) + tid * 16);
    // V tile: 64 rows (channels) x 64B, row stride N_*2 bytes in global
    async_b128((unsigned)(size_t)&Vlds[ibuf][0] + vrow * 64 + vseg * 16,
               (const char*)(vbase + (size_t)vrow * N_ + kb) + vseg * 16);
  };

  stage(0, 0);
  for (int kb = 0; kb < N_; kb += 32) {
    const int ibuf = (kb >> 5) & 1;
    if (kb + 32 < N_) {
      stage(kb + 32, ibuf ^ 1);
      // async loads complete in order per wave: <=2 outstanding => current done
      asm volatile("s_wait_asynccnt 0x2" ::: "memory");
    } else {
      asm volatile("s_wait_asynccnt 0x0" ::: "memory");
    }
    __syncthreads();  // tile visible to all waves

    const _Float16* Kt = &Klds[ibuf][0];
    const _Float16* Vt = &Vlds[ibuf][0];

    // S tiles: rows = queries, cols = keys (from LDS)
    v8f S0{}, S1{};
    S0 = wmma32(qa0, load_frag_b(Kt, 64), S0);
    S0 = wmma32(qa1, load_frag_b(Kt + 32, 64), S0);
    S1 = wmma32(qa0, load_frag_b(Kt + 16 * 64, 64), S1);
    S1 = wmma32(qa1, load_frag_b(Kt + 16 * 64 + 32, 64), S1);

    // online softmax per row (cross-lane reduce within 16-lane half-groups)
    _Float16* Pw = &Plds[wid][0];
#pragma unroll
    for (int r = 0; r < 8; ++r) {
      float t = fmaxf(S0[r], S1[r]);
      t = fmaxf(t, __shfl_xor(t, 1, 32));
      t = fmaxf(t, __shfl_xor(t, 2, 32));
      t = fmaxf(t, __shfl_xor(t, 4, 32));
      t = fmaxf(t, __shfl_xor(t, 8, 32));
      const float mn = fmaxf(mrow[r], t);
      const float sc = __expf(mrow[r] - mn);
      const float p0 = __expf(S0[r] - mn);
      const float p1 = __expf(S1[r] - mn);
      float s = p0 + p1;
      s += __shfl_xor(s, 1, 32);
      s += __shfl_xor(s, 2, 32);
      s += __shfl_xor(s, 4, 32);
      s += __shfl_xor(s, 8, 32);
      lrow[r] = lrow[r] * sc + s;
      mrow[r] = mn;
      O0[r] *= sc; O1[r] *= sc; O2[r] *= sc; O3[r] *= sc;
      const int row = r + 8 * hi;
      Pw[row * 32 + col]      = (_Float16)p0;
      Pw[row * 32 + 16 + col] = (_Float16)p1;
    }
    // wave-local D->A relayout: LDS is in-order per wave; stop compiler reordering
    asm volatile("" ::: "memory");

    const v16h pa = load_frag_a(Pw, 32);
    O0 = wmma32(pa, load_frag_b(Vt + 0 * 32, 32), O0);
    O1 = wmma32(pa, load_frag_b(Vt + 16 * 32, 32), O1);
    O2 = wmma32(pa, load_frag_b(Vt + 32 * 32, 32), O2);
    O3 = wmma32(pa, load_frag_b(Vt + 48 * 32, 32), O3);
    __syncthreads();  // all waves done with tile before next async overwrite
  }

  _Float16* ob = out1nc + ((size_t)b * N_ + q0) * C_;
#pragma unroll
  for (int r = 0; r < 8; ++r) {
    const float inv = 1.f / lrow[r];
    const int row = r + 8 * hi;
    ob[(size_t)row * C_ + 0  + col] = (_Float16)(O0[r] * inv);
    ob[(size_t)row * C_ + 16 + col] = (_Float16)(O1[r] * inv);
    ob[(size_t)row * C_ + 32 + col] = (_Float16)(O2[r] * inv);
    ob[(size_t)row * C_ + 48 + col] = (_Float16)(O3[r] * inv);
  }
}

// ---------------- Kernel 4a: channel-attention logits (64x64, reduce N) ------
__global__ __launch_bounds__(32)
void alfe_chan_logits(const _Float16* __restrict__ q2cn,  // (B,64,N)
                      const _Float16* __restrict__ k2cn,  // (B,64,N)
                      float* __restrict__ qk2) {          // (B,64,64)
  const int b = blockIdx.z, ct = blockIdx.x, dt = blockIdx.y;
  const _Float16* qb = q2cn + ((size_t)b * C_ + ct * 16) * (size_t)N_;
  const _Float16* kb = k2cn + ((size_t)b * C_ + dt * 16) * (size_t)N_;
  v8f acc{};
  for (int n = 0; n < N_; n += 32)
    acc = wmma32(load_frag_a(qb + n, N_), load_frag_b(kb + n, N_), acc);
  const int l = threadIdx.x & 31, hi = l >> 4, col = l & 15;
#pragma unroll
  for (int r = 0; r < 8; ++r)
    qk2[((size_t)b * C_ + ct * 16 + r + 8 * hi) * C_ + dt * 16 + col] = acc[r];
}

// ---------------- Kernel 4b: softmax over 64-wide rows -----------------------
__global__ void alfe_chan_softmax(const float* __restrict__ qk2,
                                  _Float16* __restrict__ qk2h) {
  const int row = blockIdx.x * blockDim.x + threadIdx.x;  // B*64 rows
  if (row >= B_ * C_) return;
  const float* p = qk2 + (size_t)row * C_;
  float mx = -1e30f;
  for (int i = 0; i < C_; ++i) mx = fmaxf(mx, p[i]);
  float s = 0.f;
  for (int i = 0; i < C_; ++i) s += __expf(p[i] - mx);
  const float inv = 1.f / s;
  for (int i = 0; i < C_; ++i)
    qk2h[(size_t)row * C_ + i] = (_Float16)(__expf(p[i] - mx) * inv);
}

// ---------------- Kernel 4c: out2 = qk2 @ V via WMMA -------------------------
__global__ __launch_bounds__(32)
void alfe_chan_apply(const _Float16* __restrict__ qk2h,  // (B,64,64)
                     const _Float16* __restrict__ vnc,   // (B,N,64) = V^T
                     _Float16* __restrict__ out2nc) {    // (B,N,64)
  const int b = blockIdx.y;
  const int n0 = blockIdx.x * 16;
  const _Float16* vb = vnc + ((size_t)b * N_ + n0) * C_;
  const v16h bf0 = load_frag_b(vb, C_);        // d = 0..31
  const v16h bf1 = load_frag_b(vb + 32, C_);   // d = 32..63
  const _Float16* qb = qk2h + (size_t)b * C_ * C_;
  const int l = threadIdx.x & 31, hi = l >> 4, col = l & 15;
#pragma unroll
  for (int ct = 0; ct < 4; ++ct) {
    v8f acc{};
    acc = wmma32(load_frag_a(qb + ct * 16 * C_, C_), bf0, acc);
    acc = wmma32(load_frag_a(qb + ct * 16 * C_ + 32, C_), bf1, acc);
#pragma unroll
    for (int r = 0; r < 8; ++r) {
      const int c = ct * 16 + r + 8 * hi;
      const int n = n0 + col;
      out2nc[((size_t)b * N_ + n) * C_ + c] = (_Float16)acc[r];
    }
  }
}

// ---------------- Kernel 5: final (64 x 128) conv via WMMA -------------------
__global__ __launch_bounds__(32)
void alfe_final_conv(const _Float16* __restrict__ w1h,    // (64,128)
                     const float* __restrict__ b1,
                     const _Float16* __restrict__ out1nc, // (B,N,64)
                     const _Float16* __restrict__ out2nc, // (B,N,64)
                     float* __restrict__ conv) {          // (B,64,N)
  const int b = blockIdx.y;
  const int n0 = blockIdx.x * 16;
  const _Float16* o1 = out1nc + ((size_t)b * N_ + n0) * C_;
  const _Float16* o2 = out2nc + ((size_t)b * N_ + n0) * C_;
  const v16h bf0 = load_frag_b(o1, C_);       // k 0..31
  const v16h bf1 = load_frag_b(o1 + 32, C_);  // k 32..63
  const v16h bf2 = load_frag_b(o2, C_);       // k 64..95
  const v16h bf3 = load_frag_b(o2 + 32, C_);  // k 96..127
  const int l = threadIdx.x & 31, hi = l >> 4, col = l & 15;
#pragma unroll
  for (int ct = 0; ct < 4; ++ct) {
    const _Float16* wr = w1h + (size_t)ct * 16 * 128;
    v8f acc{};
    acc = wmma32(load_frag_a(wr, 128), bf0, acc);
    acc = wmma32(load_frag_a(wr + 32, 128), bf1, acc);
    acc = wmma32(load_frag_a(wr + 64, 128), bf2, acc);
    acc = wmma32(load_frag_a(wr + 96, 128), bf3, acc);
#pragma unroll
    for (int r = 0; r < 8; ++r) {
      const int cout = ct * 16 + r + 8 * hi;
      conv[((size_t)b * C_ + cout) * (size_t)N_ + n0 + col] = acc[r] + b1[cout];
    }
  }
}

// ---------------- Kernel 6: bilinear x2 upsample (half-pixel) + residual -----
__global__ void alfe_upsample_add(const float* __restrict__ conv,  // (B,64,HH,WH)
                                  const float* __restrict__ x,
                                  float* __restrict__ out) {
  const size_t i = (size_t)blockIdx.x * 256 + threadIdx.x;
  const size_t total = (size_t)B_ * C_ * H_ * W_;
  if (i >= total) return;
  const int w = (int)(i & (W_ - 1));
  const int h = (int)((i >> 7) & (H_ - 1));
  const int bc = (int)(i >> 14);
  const float sh = h * 0.5f - 0.25f;
  const float sw = w * 0.5f - 0.25f;
  const int h0 = (int)floorf(sh), w0 = (int)floorf(sw);
  const float fh = sh - (float)h0, fw = sw - (float)w0;
  const int h0c = min(max(h0, 0), HH_ - 1), h1c = min(max(h0 + 1, 0), HH_ - 1);
  const int w0c = min(max(w0, 0), WH_ - 1), w1c = min(max(w0 + 1, 0), WH_ - 1);
  const float* cb = conv + (size_t)bc * N_;
  const float v00 = cb[h0c * WH_ + w0c], v01 = cb[h0c * WH_ + w1c];
  const float v10 = cb[h1c * WH_ + w0c], v11 = cb[h1c * WH_ + w1c];
  const float v = (1.f - fh) * ((1.f - fw) * v00 + fw * v01) +
                  fh * ((1.f - fw) * v10 + fw * v11);
  out[i] = v + x[i];
}

// ---------------- Host launch ----------------
extern "C" void kernel_launch(void* const* d_in, const int* in_sizes, int n_in,
                              void* d_out, int out_size, void* d_ws, size_t ws_size,
                              hipStream_t stream) {
  (void)in_sizes; (void)n_in; (void)out_size; (void)ws_size;
  const float* x        = (const float*)d_in[0];
  const float* pm_gamma = (const float*)d_in[1];
  const float* pm_beta  = (const float*)d_in[2];
  const float* pm_w     = (const float*)d_in[3];
  const float* pm_b     = (const float*)d_in[4];
  const float* w1 = (const float*)d_in[5];  const float* b1 = (const float*)d_in[6];
  const float* w2 = (const float*)d_in[7];  const float* b2 = (const float*)d_in[8];
  const float* w3 = (const float*)d_in[9];  const float* b3 = (const float*)d_in[10];
  const float* w4 = (const float*)d_in[11]; const float* b4 = (const float*)d_in[12];
  const float* w5 = (const float*)d_in[13]; const float* b5 = (const float*)d_in[14];
  const float* w6 = (const float*)d_in[15]; const float* b6 = (const float*)d_in[16];
  float* out = (float*)d_out;

  // workspace layout (~22.2 MB, 256B aligned slabs)
  char* p = (char*)d_ws;
  auto take = [&p](size_t bytes) -> char* {
    char* r = p; p += (bytes + 255) & ~(size_t)255; return r;
  };
  const size_t h_bnc = (size_t)B_ * N_ * C_ * sizeof(_Float16);  // 2 MB each
  _Float16* c1h    = (_Float16*)take(h_bnc);
  _Float16* qnc    = (_Float16*)take(h_bnc);
  _Float16* knc    = (_Float16*)take(h_bnc);
  _Float16* vcn    = (_Float16*)take(h_bnc);
  _Float16* vnc    = (_Float16*)take(h_bnc);
  _Float16* q2cn   = (_Float16*)take(h_bnc);
  _Float16* k2cn   = (_Float16*)take(h_bnc);
  _Float16* out1nc = (_Float16*)take(h_bnc);
  _Float16* out2nc = (_Float16*)take(h_bnc);
  float*    qk2    = (float*)take((size_t)B_ * C_ * C_ * sizeof(float));
  _Float16* qk2h   = (_Float16*)take((size_t)B_ * C_ * C_ * sizeof(_Float16));
  float*    convb  = (float*)take((size_t)B_ * C_ * N_ * sizeof(float));   // 4 MB
  _Float16* wh5    = (_Float16*)take((size_t)5 * 4096 * sizeof(_Float16));
  _Float16* w1h    = (_Float16*)take((size_t)8192 * sizeof(_Float16));

  alfe_cvt_weights<<<32, 256, 0, stream>>>(w2, w3, w4, w5, w6, w1, wh5, w1h);

  alfe_unshuffle_ln_proj<<<B_ * (N_ / 32), 256, 0, stream>>>(
      x, pm_gamma, pm_beta, pm_w, pm_b, c1h);

  alfe_qkv_proj<<<dim3(N_ / 16, B_), 32, 0, stream>>>(
      c1h, wh5, b2, b3, b4, b5, b6, qnc, knc, vcn, vnc, q2cn, k2cn);

  alfe_flash_attn<<<dim3(N_ / (16 * FA_WAVES), B_), 32 * FA_WAVES, 0, stream>>>(
      qnc, knc, vcn, out1nc);

  alfe_chan_logits<<<dim3(4, 4, B_), 32, 0, stream>>>(q2cn, k2cn, qk2);
  alfe_chan_softmax<<<1, 256, 0, stream>>>(qk2, qk2h);
  alfe_chan_apply<<<dim3(N_ / 16, B_), 32, 0, stream>>>(qk2h, vnc, out2nc);

  alfe_final_conv<<<dim3(N_ / 16, B_), 32, 0, stream>>>(
      w1h, b1, out1nc, out2nc, convb);

  const size_t total = (size_t)B_ * C_ * H_ * W_;
  alfe_upsample_add<<<(unsigned)((total + 255) / 256), 256, 0, stream>>>(
      convb, x, out);
}
